// GCN2_LSTM_Model_52604759441753
// MI455X (gfx1250) — matrix operand order repaired and checked by reading, not compiled, not run on hardware
//
#include <hip/hip_runtime.h>
#include <stdint.h>

// ---------------------------------------------------------------------------
// Types for CDNA5 WMMA (gfx1250, wave32)
// ---------------------------------------------------------------------------
typedef __attribute__((ext_vector_type(16))) __bf16 v16bf;
typedef __attribute__((ext_vector_type(8)))  float  v8f;

__device__ __forceinline__ unsigned short f32_to_bf16(float f) {
    union { float f; uint32_t u; } v; v.f = f;
    uint32_t r = v.u + 0x7FFFu + ((v.u >> 16) & 1u);   // round-to-nearest-even
    return (unsigned short)(r >> 16);
}

union FragU { uint4 q[2]; v16bf v; };

// A-fragment (16x32 bf16 tile, row-major in LDS):
// lane (lo,hi): row M=lo ; elements 0..7 -> K = hi*8 + e ; 8..15 -> K = 16 + hi*8 + (e-8)
__device__ __forceinline__ v16bf load_frag_a(const unsigned short* rowBase, int hi) {
    FragU f;
    f.q[0] = *(const uint4*)(rowBase + hi * 8);
    f.q[1] = *(const uint4*)(rowBase + 16 + hi * 8);
    return f.v;
}
// B-fragment (columns stored as rows in LDS, i.e. Bs[n][k]):
// lane (lo,hi): col N=lo ; elements 0..7 -> K = hi*8 + e ; 8..15 -> K = 16 + hi*8 + (e-8)
__device__ __forceinline__ v16bf load_frag_b(const unsigned short* rowBase, int hi) {
    FragU f;
    f.q[0] = *(const uint4*)(rowBase + hi * 8);
    f.q[1] = *(const uint4*)(rowBase + 16 + hi * 8);
    return f.v;
}

// CDNA5 async global->LDS copy: 16 bytes per active lane, tracked by ASYNCcnt.
// (GLOBAL_LOAD_ASYNC_TO_LDS_B128, VGLOBAL opcode 98 — bypasses VGPR staging.)
__device__ __forceinline__ void async_copy_b128(void* lds_dst, const void* gsrc) {
    unsigned lds = (unsigned)(unsigned long long)lds_dst;   // addrspace(3) byte offset
    asm volatile("global_load_async_to_lds_b128 %0, %1, off"
                 :: "v"(lds), "v"((unsigned long long)gsrc)
                 : "memory");
}
__device__ __forceinline__ void async_wait0() {
    asm volatile("s_wait_asynccnt 0x0" ::: "memory");
}

// ---------------------------------------------------------------------------
// Generic tiled bf16 WMMA GEMM:  C[z] = A[z] (MxK) * B[z] (KxN or NxK^T) (+bias / +Cinit)
//   - A row-major MxK (f32 or bf16-u16), lda = K
//   - B: if bTrans: row-major NxK (computes A*B^T); else row-major KxN
//   - C: optional f32 out (ldc=N) and/or bf16 out; batch z offsets via strides/mods
//   - Cinit: optional f32 accumulator init, Cinit[m*ldci + n] (single batch use)
// Block: 64x64 tile, BK=32, 256 threads = 8 waves (4 m-bands x 2 n-bands),
// each wave -> one 16-row band, two 16-col WMMA tiles.
// bf16 operands stage through GLOBAL_LOAD_ASYNC_TO_LDS_B128; f32 operands are
// converted to bf16 in VALU on the way to LDS.
// ---------------------------------------------------------------------------
__global__ __launch_bounds__(256)
void gemm_bf16_wmma(const void* Ap, int aF32, long long aStride, int aMod,
                    const void* Bp, int bF32, int bTrans, long long bStride, int bMod,
                    float* Cf, unsigned short* Cb, long long cStride,
                    const float* bias, long long biasStride, int biasMod,
                    const float* Cinit, int ldci,
                    int Mdim, int Ndim, int Kdim)
{
    __shared__ unsigned short As[64][40];   // 40 = 32 + 8 pad (80B row stride, 16B aligned)
    __shared__ unsigned short Bs[64][40];   // stored n-major: Bs[n][k]

    const int tid = threadIdx.x;
    const int z   = blockIdx.z;
    const int m0  = blockIdx.y * 64;
    const int n0  = blockIdx.x * 64;

    const long long aOff = (long long)(aMod ? (z % aMod) : z) * aStride;
    const long long bOff = (long long)(bMod ? (z % bMod) : z) * bStride;

    const int lane = tid & 31;
    const int wv   = tid >> 5;       // 0..7
    const int wm   = wv & 3;         // m band 0..3
    const int wn   = wv >> 2;        // n band 0..1
    const int lo   = lane & 15;
    const int hi   = lane >> 4;

    v8f acc0, acc1;
    {
        const int   mBase = m0 + wm * 16 + hi * 8;
        const int   nb0   = n0 + wn * 32 + lo;
        const int   nb1   = nb0 + 16;
        const float* bpv = nullptr;
        if (bias) bpv = bias + (long long)(biasMod ? (z % biasMod) : z) * biasStride;
#pragma unroll
        for (int j = 0; j < 8; ++j) {
            int m = mBase + j;
            float v0 = 0.f, v1 = 0.f;
            if (Cinit) {
                if (m < Mdim) {
                    if (nb0 < Ndim) v0 = Cinit[(long long)m * ldci + nb0];
                    if (nb1 < Ndim) v1 = Cinit[(long long)m * ldci + nb1];
                }
            } else if (bpv) {
                if (nb0 < Ndim) v0 = bpv[nb0];
                if (nb1 < Ndim) v1 = bpv[nb1];
            }
            acc0[j] = v0; acc1[j] = v1;
        }
    }

    const int arow = tid >> 2;          // 0..63
    const int aseg = (tid & 3) * 8;     // 0,8,16,24
    const int bk   = tid & 31;          // 0..31 (no-trans B loader)
    const int bns  = (tid >> 5) * 8;    // 0..56

    for (int k0 = 0; k0 < Kdim; k0 += 32) {
        // ---- stage A tile (64x32) ----
        {
            int gm = m0 + arow;
            unsigned short* dst = &As[arow][aseg];
            if (gm < Mdim) {
                long long base = aOff + (long long)gm * Kdim + (k0 + aseg);
                if (aF32) {
                    const float* src = (const float*)Ap + base;
#pragma unroll
                    for (int j = 0; j < 8; ++j) dst[j] = f32_to_bf16(src[j]);
                } else {
                    async_copy_b128(dst, (const unsigned short*)Ap + base);
                }
            } else {
                *(uint4*)dst = make_uint4(0u, 0u, 0u, 0u);
            }
        }
        // ---- stage B tile into Bs[n][k] ----
        if (bTrans) {
            int gn = n0 + arow;
            unsigned short* dst = &Bs[arow][aseg];
            if (gn < Ndim) {
                long long base = bOff + (long long)gn * Kdim + (k0 + aseg);
                if (bF32) {
                    const float* src = (const float*)Bp + base;
#pragma unroll
                    for (int j = 0; j < 8; ++j) dst[j] = f32_to_bf16(src[j]);
                } else {
                    async_copy_b128(dst, (const unsigned short*)Bp + base);
                }
            } else {
                *(uint4*)dst = make_uint4(0u, 0u, 0u, 0u);
            }
        } else {
            int gk  = k0 + bk;
            int gnb = n0 + bns;
            if (bF32) {
                const float* src = (const float*)Bp + bOff + (long long)gk * Ndim + gnb;
#pragma unroll
                for (int j = 0; j < 8; ++j)
                    Bs[bns + j][bk] = (gnb + j < Ndim) ? f32_to_bf16(src[j]) : (unsigned short)0;
            } else {
                const unsigned short* src = (const unsigned short*)Bp + bOff + (long long)gk * Ndim + gnb;
#pragma unroll
                for (int j = 0; j < 8; ++j)
                    Bs[bns + j][bk] = (gnb + j < Ndim) ? src[j] : (unsigned short)0;
            }
        }
        async_wait0();          // drain ASYNCcnt before waves read each other's tiles
        __syncthreads();

        v16bf a  = load_frag_a(&As[wm * 16 + lo][0], hi);
        v16bf b0 = load_frag_b(&Bs[wn * 32 + lo][0], hi);
        v16bf b1 = load_frag_b(&Bs[wn * 32 + 16 + lo][0], hi);
        acc0 = __builtin_amdgcn_wmma_f32_16x16x32_bf16(false, a, false, b0, (short)0, acc0, false, false);
        acc1 = __builtin_amdgcn_wmma_f32_16x16x32_bf16(false, a, false, b1, (short)0, acc1, false, false);
        __syncthreads();
    }

    // ---- store ----
    {
        const long long cOff = (long long)z * cStride;
        const int mBase = m0 + wm * 16 + hi * 8;
        const int nb0   = n0 + wn * 32 + lo;
        const int nb1   = nb0 + 16;
#pragma unroll
        for (int j = 0; j < 8; ++j) {
            int m = mBase + j;
            if (m < Mdim) {
                if (nb0 < Ndim) {
                    long long o = cOff + (long long)m * Ndim + nb0;
                    if (Cf) Cf[o] = acc0[j];
                    if (Cb) Cb[o] = f32_to_bf16(acc0[j]);
                }
                if (nb1 < Ndim) {
                    long long o = cOff + (long long)m * Ndim + nb1;
                    if (Cf) Cf[o] = acc1[j];
                    if (Cb) Cb[o] = f32_to_bf16(acc1[j]);
                }
            }
        }
    }
}

// ---------------------------------------------------------------------------
// GCN normalization:  deg[i] = 1 + sum_j adj[j,i];  Mt[n,j] = d_j*(adj[j,n]+[j==n])*d_n
// ---------------------------------------------------------------------------
__global__ void degree_kernel(const float* adj, float* dinv, int n) {
    int i = blockIdx.x * blockDim.x + threadIdx.x;
    if (i >= n) return;
    float s = 1.0f;                       // self loop weight
    for (int j = 0; j < n; ++j) s += adj[(long long)j * n + i];
    dinv[i] = rsqrtf(s);
}

__global__ void norm_adj_kernel(const float* adj, const float* dinv, unsigned short* Mt, int n) {
    long long idx = (long long)blockIdx.x * blockDim.x + threadIdx.x;
    if (idx >= (long long)n * n) return;
    int nr = (int)(idx / n), j = (int)(idx % n);
    float a = adj[(long long)j * n + nr] + (j == nr ? 1.f : 0.f);
    Mt[(long long)nr * n + j] = f32_to_bf16(dinv[j] * a * dinv[nr]);
}

// ---------------------------------------------------------------------------
// GraphNorm (stats over B=16, leading axis) + residual + ReLU, in place on h,
// also emits bf16 copy for the following GEMM. All B samples kept in registers.
// ---------------------------------------------------------------------------
__global__ void graphnorm_kernel(float* h, const float* resid,
                                 const float* w, const float* bb, const float* ms,
                                 unsigned short* out_bf, long long plane, int Hdim)
{
    long long idx = (long long)blockIdx.x * blockDim.x + threadIdx.x;
    if (idx >= plane) return;
    const int ch = (int)(idx % Hdim);
    float v[16];
    float mean = 0.f;
#pragma unroll
    for (int s = 0; s < 16; ++s) { v[s] = h[(long long)s * plane + idx]; mean += v[s]; }
    mean *= (1.0f / 16.0f);
    const float m = ms[ch] * mean;
    float var = 0.f;
#pragma unroll
    for (int s = 0; s < 16; ++s) { v[s] -= m; var += v[s] * v[s]; }
    var *= (1.0f / 16.0f);
    const float inv = rsqrtf(var + 1e-5f);
    const float wc = w[ch], bc = bb[ch];
#pragma unroll
    for (int s = 0; s < 16; ++s) {
        float r = wc * v[s] * inv + bc + resid[(long long)s * plane + idx];
        r = fmaxf(r, 0.f);
        h[(long long)s * plane + idx] = r;
        out_bf[(long long)s * plane + idx] = f32_to_bf16(r);
    }
}

// ---------------------------------------------------------------------------
// LSTM cell pointwise update (gate order i,f,g,o)
// ---------------------------------------------------------------------------
__global__ void lstm_cell_kernel(const float* g, float* c, unsigned short* hbf,
                                 float* y, int t, int T, int HL)
{
    int idx = blockIdx.x * blockDim.x + threadIdx.x;
    if (idx >= 16 * HL) return;
    int bi = idx / HL, u = idx % HL;
    const float* gb = g + (long long)bi * 4 * HL;
    float gi = gb[u], gf = gb[HL + u], gg = gb[2 * HL + u], go = gb[3 * HL + u];
    float si = 1.f / (1.f + __expf(-gi));
    float sf = 1.f / (1.f + __expf(-gf));
    float so = 1.f / (1.f + __expf(-go));
    float cv = sf * c[idx] + si * tanhf(gg);
    float hv = so * tanhf(cv);
    c[idx] = cv;
    hbf[idx] = f32_to_bf16(hv);
    y[((long long)bi * T + t) * HL + u] = hv;
}

__global__ void add_vec_kernel(const float* a, const float* b, float* o, int n) {
    int i = blockIdx.x * blockDim.x + threadIdx.x;
    if (i < n) o[i] = a[i] + b[i];
}

__global__ void fill_zero_kernel(unsigned int* p, long long n) {
    long long i = (long long)blockIdx.x * blockDim.x + threadIdx.x;
    long long s = (long long)gridDim.x * blockDim.x;
    for (; i < n; i += s) p[i] = 0u;
}

__global__ void cvt_bf16_kernel(const float* in, unsigned short* out, long long n) {
    long long i = (long long)blockIdx.x * blockDim.x + threadIdx.x;
    long long s = (long long)gridDim.x * blockDim.x;
    for (; i < n; i += s) out[i] = f32_to_bf16(in[i]);
}

__global__ void proj_kernel(const float* y1, const float* Wp, const float* bp,
                            float* out, int T, int HL, int O)
{
    int idx = blockIdx.x * blockDim.x + threadIdx.x;
    if (idx >= 16 * O) return;
    int bi = idx / O, o = idx % O;
    const float* yr = y1 + ((long long)bi * T + (T - 1)) * HL;
    const float* wr = Wp + (long long)o * HL;
    float s = bp[o];
    for (int k = 0; k < HL; ++k) s += yr[k] * wr[k];
    out[(long long)bi * O + o] = s;
}

// ---------------------------------------------------------------------------
extern "C" void kernel_launch(void* const* d_in, const int* in_sizes, int n_in,
                              void* d_out, int out_size, void* d_ws, size_t ws_size,
                              hipStream_t stream)
{
    (void)in_sizes; (void)n_in; (void)out_size; (void)ws_size;

    const int Bsz = 16, T = 32, Nn = 512, C = 128, H = 128, HL = 512, O = 8;
    const int BT = Bsz * T;                 // 512
    const long long plane = (long long)T * Nn * H;   // 2,097,152

    const float* x    = (const float*)d_in[0];
    const float* adj  = (const float*)d_in[1];
    const float* W1   = (const float*)d_in[2];
    const float* b1   = (const float*)d_in[3];
    const float* W2   = (const float*)d_in[4];
    const float* b2   = (const float*)d_in[5];
    const float* gn1w = (const float*)d_in[6];
    const float* gn1b = (const float*)d_in[7];
    const float* gn1m = (const float*)d_in[8];
    const float* gn2w = (const float*)d_in[9];
    const float* gn2b = (const float*)d_in[10];
    const float* gn2m = (const float*)d_in[11];
    const float* Wih0 = (const float*)d_in[12];
    const float* Whh0 = (const float*)d_in[13];
    const float* bih0 = (const float*)d_in[14];
    const float* bhh0 = (const float*)d_in[15];
    const float* Wih1 = (const float*)d_in[16];
    const float* Whh1 = (const float*)d_in[17];
    const float* bih1 = (const float*)d_in[18];
    const float* bhh1 = (const float*)d_in[19];
    const float* Wp   = (const float*)d_in[20];
    const float* bp   = (const float*)d_in[21];

    // ---- workspace bump allocator ----
    char* ws = (char*)d_ws;
    size_t off = 0;
    auto alloc = [&](size_t bytes) -> void* {
        off = (off + 255) & ~(size_t)255;
        void* p = ws + off;
        off += bytes;
        return p;
    };

    float*          dinv   = (float*)alloc((size_t)Nn * 4);
    unsigned short* Mt     = (unsigned short*)alloc((size_t)Nn * Nn * 2);
    unsigned short* hw_bf  = (unsigned short*)alloc((size_t)BT * Nn * H * 2);  // reused layer2
    float*          h1     = (float*)alloc((size_t)BT * Nn * H * 4);
    unsigned short* h1n_bf = (unsigned short*)alloc((size_t)BT * Nn * H * 2);  // reused as seq bf16
    float*          h2     = (float*)alloc((size_t)BT * Nn * H * 4);
    float*          xpart0 = (float*)alloc((size_t)BT * 4 * HL * 4);
    float*          xpart1 = (float*)alloc((size_t)BT * 4 * HL * 4);
    float*          bsum0  = (float*)alloc((size_t)4 * HL * 4);
    float*          bsum1  = (float*)alloc((size_t)4 * HL * 4);
    float*          gbuf   = (float*)alloc((size_t)Bsz * 4 * HL * 4);
    float*          cbuf   = (float*)alloc((size_t)Bsz * HL * 4);
    unsigned short* hbf    = (unsigned short*)alloc((size_t)Bsz * HL * 2);
    float*          y0     = (float*)alloc((size_t)BT * HL * 4);
    unsigned short* y0bf   = (unsigned short*)alloc((size_t)BT * HL * 2);
    float*          y1     = (float*)alloc((size_t)BT * HL * 4);

    // ---- 1) adjacency normalization (Mt[n,j] bf16) ----
    degree_kernel<<<(Nn + 255) / 256, 256, 0, stream>>>(adj, dinv, Nn);
    norm_adj_kernel<<<(Nn * Nn + 255) / 256, 256, 0, stream>>>(adj, dinv, Mt, Nn);

    // ---- 2) GCN layer 1 ----
    // hw = x[b,t] (512x128 f32) @ W1[t] (128x128 f32) -> bf16
    gemm_bf16_wmma<<<dim3(H / 64, Nn / 64, BT), 256, 0, stream>>>(
        x, 1, (long long)Nn * C, 0,
        W1, 1, 0, (long long)C * H, T,
        nullptr, hw_bf, (long long)Nn * H,
        nullptr, 0, 1, nullptr, 0,
        Nn, H, C);
    // h1 = Mt (512x512 bf16, async-LDS path) @ hw (512x128 bf16) + b1[t]
    gemm_bf16_wmma<<<dim3(H / 64, Nn / 64, BT), 256, 0, stream>>>(
        Mt, 0, 0, 1,
        hw_bf, 0, 0, (long long)Nn * H, 0,
        h1, nullptr, (long long)Nn * H,
        b1, H, T, nullptr, 0,
        Nn, H, Nn);
    // GraphNorm1 + residual(x) + ReLU  (in place, bf16 copy out)
    graphnorm_kernel<<<(unsigned)((plane + 255) / 256), 256, 0, stream>>>(
        h1, x, gn1w, gn1b, gn1m, h1n_bf, plane, H);

    // ---- 3) GCN layer 2 ----
    gemm_bf16_wmma<<<dim3(H / 64, Nn / 64, BT), 256, 0, stream>>>(
        h1n_bf, 0, (long long)Nn * H, 0,
        W2, 1, 0, (long long)H * H, T,
        nullptr, hw_bf, (long long)Nn * H,
        nullptr, 0, 1, nullptr, 0,
        Nn, H, H);
    gemm_bf16_wmma<<<dim3(H / 64, Nn / 64, BT), 256, 0, stream>>>(
        Mt, 0, 0, 1,
        hw_bf, 0, 0, (long long)Nn * H, 0,
        h2, nullptr, (long long)Nn * H,
        b2, H, T, nullptr, 0,
        Nn, H, Nn);
    // GraphNorm2 + residual(h1) + ReLU -> seq bf16 (reuses h1n_bf storage)
    unsigned short* seqbf = h1n_bf;
    graphnorm_kernel<<<(unsigned)((plane + 255) / 256), 256, 0, stream>>>(
        h2, h1, gn2w, gn2b, gn2m, seqbf, plane, H);

    // ---- 4) LSTM input projections ----
    add_vec_kernel<<<(4 * HL + 255) / 256, 256, 0, stream>>>(bih0, bhh0, bsum0, 4 * HL);
    add_vec_kernel<<<(4 * HL + 255) / 256, 256, 0, stream>>>(bih1, bhh1, bsum1, 4 * HL);

    // xpart0 = seq[512,65536] @ Wih0^T (f32 NxK, converted on the fly) + bsum0
    gemm_bf16_wmma<<<dim3(4 * HL / 64, BT / 64, 1), 256, 0, stream>>>(
        seqbf, 0, 0, 1,
        Wih0, 1, 1, 0, 1,
        xpart0, nullptr, 0,
        bsum0, 0, 1, nullptr, 0,
        BT, 4 * HL, Nn * H);

    // ---- 5) LSTM layer 0 recurrence ----
    fill_zero_kernel<<<64, 256, 0, stream>>>((unsigned int*)cbuf, (long long)Bsz * HL);
    fill_zero_kernel<<<64, 256, 0, stream>>>((unsigned int*)hbf, (long long)Bsz * HL / 2);
    for (int t = 0; t < T; ++t) {
        // g = xpart0[:,t,:] + h @ Whh0^T   (M=16, N=2048, K=512)
        gemm_bf16_wmma<<<dim3(4 * HL / 64, 1, 1), 256, 0, stream>>>(
            hbf, 0, 0, 1,
            Whh0, 1, 1, 0, 1,
            gbuf, nullptr, 0,
            nullptr, 0, 1,
            xpart0 + (long long)t * 4 * HL, T * 4 * HL,
            Bsz, 4 * HL, HL);
        lstm_cell_kernel<<<(Bsz * HL + 255) / 256, 256, 0, stream>>>(gbuf, cbuf, hbf, y0, t, T, HL);
    }

    // ---- 6) LSTM layer 1 ----
    cvt_bf16_kernel<<<256, 256, 0, stream>>>(y0, y0bf, (long long)BT * HL);
    gemm_bf16_wmma<<<dim3(4 * HL / 64, BT / 64, 1), 256, 0, stream>>>(
        y0bf, 0, 0, 1,
        Wih1, 1, 1, 0, 1,
        xpart1, nullptr, 0,
        bsum1, 0, 1, nullptr, 0,
        BT, 4 * HL, HL);
    fill_zero_kernel<<<64, 256, 0, stream>>>((unsigned int*)cbuf, (long long)Bsz * HL);
    fill_zero_kernel<<<64, 256, 0, stream>>>((unsigned int*)hbf, (long long)Bsz * HL / 2);
    for (int t = 0; t < T; ++t) {
        gemm_bf16_wmma<<<dim3(4 * HL / 64, 1, 1), 256, 0, stream>>>(
            hbf, 0, 0, 1,
            Whh1, 1, 1, 0, 1,
            gbuf, nullptr, 0,
            nullptr, 0, 1,
            xpart1 + (long long)t * 4 * HL, T * 4 * HL,
            Bsz, 4 * HL, HL);
        lstm_cell_kernel<<<(Bsz * HL + 255) / 256, 256, 0, stream>>>(gbuf, cbuf, hbf, y1, t, T, HL);
    }

    // ---- 7) output projection ----
    proj_kernel<<<1, 128, 0, stream>>>(y1, Wp, bp, (float*)d_out, T, HL, O);
}
